// StokenAttention_88046829568007
// MI455X (gfx1250) — matrix operand order repaired and checked by reading, not compiled, not run on hardware
//
#include <hip/hip_runtime.h>
#include <math.h>

typedef __attribute__((ext_vector_type(16))) _Float16 v16h;
typedef __attribute__((ext_vector_type(8)))  float    v8f;

#define WMMA_F32_F16(a,b,c) \
  __builtin_amdgcn_wmma_f32_16x16x32_f16(false,(a),false,(b),(short)0,(c),false,false)

// ---- geometry ----
#define BB    8
#define CC    256
#define HH_   128
#define WW_   128
#define GH    16      // super-token grid height (hh)
#define GW    16      // super-token grid width (ww)
#define NS    256     // GH*GW
#define PP    64      // 8*8 pixels per super token
#define NHEAD 8
#define HD    32
#define NAtt  1024    // VN*16*16 attention sequence length

// ------------------------------------------------------------------
// WMMA fragment loaders.
// A operand: stored [M][K], K contiguous, ldk = row stride (halves).
//   lane L: row = L&15; v16h elems 0..15 = K (base + (L>=16 ? 16:0)) + e
// B operand: stored [N][K] (i.e. B transposed), K contiguous.
//   lane L: col = L&15; v16h elems 0..15 = K (base + (L>=16 ? 16:0)) + e
// ------------------------------------------------------------------
__device__ __forceinline__ v16h frag_A16(const _Float16* base, int ldk) {
  const int lane = threadIdx.x & 31;
  const _Float16* p = base + (lane & 15) * ldk + ((lane >> 4) ? 16 : 0);
  v16h r;
#pragma unroll
  for (int e = 0; e < 16; ++e) r[e] = p[e];
  return r;
}

__device__ __forceinline__ v16h frag_B16(const _Float16* base, int ldk) {
  const int lane = threadIdx.x & 31;
  const _Float16* p = base + (lane & 15) * ldk + ((lane >> 4) ? 16 : 0);
  v16h r;
#pragma unroll
  for (int e = 0; e < 16; ++e) r[e] = p[e];
  return r;
}

// ------------------------------------------------------------------
// K1: super-token mean over 8x8 blocks -> f16
// ------------------------------------------------------------------
__global__ void k_stoken_mean(const float* __restrict__ x, _Float16* __restrict__ stoken_h) {
  int t = blockIdx.x * blockDim.x + threadIdx.x;
  if (t >= BB * CC * NS) return;
  int ij = t & 255, c = (t >> 8) & 255, b = t >> 16;
  int i = ij >> 4, j = ij & 15;
  const float* px = x + ((size_t)(b * CC + c) * HH_ + i * 8) * WW_ + j * 8;
  float s = 0.f;
#pragma unroll
  for (int ph = 0; ph < 8; ++ph)
#pragma unroll
    for (int pw = 0; pw < 8; ++pw) s += px[ph * WW_ + pw];
  stoken_h[t] = (_Float16)(s * (1.0f / 64.0f));
}

// ------------------------------------------------------------------
// K2: affinity = softmax_9( pixel(64xC) @ sf(Cx9) / 16 ) per (b,n)
//     writes aff_h (k padded to 16, zeros) and per-k column sums.
// block = 128 threads (4 waves), one super-token per block.
// ------------------------------------------------------------------
__global__ void k_affinity(const float* __restrict__ x, const _Float16* __restrict__ stoken_h,
                           _Float16* __restrict__ aff_h, float* __restrict__ colsum) {
  __shared__ _Float16 pixA[PP][CC];   // [p][c]  (A operand, K=c)
  __shared__ _Float16 sfB[16][CC];    // [k][c]  (B operand [N][K])
  __shared__ float    logit[PP][16];
  const int n = blockIdx.x, b = blockIdx.y;
  const int i = n >> 4, j = n & 15;
  const int t = threadIdx.x;
  for (int c = t; c < CC; c += 128) {
    const float* px = x + ((size_t)(b * CC + c) * HH_ + i * 8) * WW_ + j * 8;
#pragma unroll
    for (int p = 0; p < PP; ++p)
      pixA[p][c] = (_Float16)px[(p >> 3) * WW_ + (p & 7)];
#pragma unroll
    for (int k = 0; k < 16; ++k) {
      _Float16 v = (_Float16)0.f;
      if (k < 9) {
        int ni = i + (k / 3) - 1, nj = j + (k % 3) - 1;
        if (ni >= 0 && ni < GH && nj >= 0 && nj < GW)
          v = stoken_h[(size_t)(b * CC + c) * NS + ni * GW + nj];
      }
      sfB[k][c] = v;
    }
  }
  __syncthreads();
  const int w = t >> 5, lane = t & 31;
  v8f acc = {};
  for (int k0 = 0; k0 < CC; k0 += 32) {
    v16h a  = frag_A16(&pixA[w * 16][k0], CC);
    v16h bf = frag_B16(&sfB[0][k0], CC);
    acc = WMMA_F32_F16(a, bf, acc);
  }
  {
    const int rowoff = w * 16 + ((lane >> 4) ? 8 : 0);
#pragma unroll
    for (int r = 0; r < 8; ++r)
      logit[rowoff + r][lane & 15] = acc[r] * 0.0625f;   // scale = C^-0.5
  }
  __syncthreads();
  if (t < PP) {
    float m = -1e30f;
#pragma unroll
    for (int k = 0; k < 9; ++k) m = fmaxf(m, logit[t][k]);
    float s = 0.f, e[9];
#pragma unroll
    for (int k = 0; k < 9; ++k) { e[k] = __expf(logit[t][k] - m); s += e[k]; }
    float inv = 1.f / s;
    _Float16* dst = aff_h + ((size_t)(b * NS + n) * PP + t) * 16;
#pragma unroll
    for (int k = 0; k < 16; ++k) {
      float a = (k < 9) ? e[k] * inv : 0.f;
      logit[t][k] = a;
      dst[k] = (_Float16)a;
    }
  }
  __syncthreads();
  if (t < 16) {
    float s = 0.f;
    for (int p = 0; p < PP; ++p) s += logit[p][t];
    colsum[(size_t)(b * NS + n) * 16 + t] = s;
  }
}

// ------------------------------------------------------------------
// K3: st_n(Cx16) = pixel^T(Cx64) @ affinity(64x16) per (b,n) -> f32 stn
// block = 256 threads (8 waves)
// ------------------------------------------------------------------
__global__ void k_stn(const float* __restrict__ x, const _Float16* __restrict__ aff_h,
                      float* __restrict__ stn) {
  __shared__ _Float16 pixT[CC][PP];   // [c][p] (A operand, K=p)
  __shared__ _Float16 affB[16][PP];   // [k][p] (B operand [N][K])
  const int n = blockIdx.x, b = blockIdx.y;
  const int i = n >> 4, j = n & 15;
  const int t = threadIdx.x;
  {
    const int c = t;
    const float* px = x + ((size_t)(b * CC + c) * HH_ + i * 8) * WW_ + j * 8;
#pragma unroll
    for (int p = 0; p < PP; ++p)
      pixT[c][p] = (_Float16)px[(p >> 3) * WW_ + (p & 7)];
  }
  if (t < PP) {
    const _Float16* src = aff_h + ((size_t)(b * NS + n) * PP + t) * 16;
#pragma unroll
    for (int k = 0; k < 16; ++k) affB[k][t] = src[k];
  }
  __syncthreads();
  const int w = t >> 5, lane = t & 31;
  float* out = stn + (size_t)(b * NS + n) * CC * 16;
  for (int mt = w * 2; mt < w * 2 + 2; ++mt) {
    v8f acc = {};
    for (int k0 = 0; k0 < PP; k0 += 32) {
      v16h a  = frag_A16(&pixT[mt * 16][k0], PP);
      v16h bf = frag_B16(&affB[0][k0], PP);
      acc = WMMA_F32_F16(a, bf, acc);
    }
    const int c0 = mt * 16 + ((lane >> 4) ? 8 : 0);
#pragma unroll
    for (int r = 0; r < 8; ++r)
      out[(size_t)(c0 + r) * 16 + (lane & 15)] = acc[r];
  }
}

// ------------------------------------------------------------------
// K4a/K4b: 3x3 fold-gather of colsum -> aff_sum, and stn -> st2 (f16)
// ------------------------------------------------------------------
__global__ void k_affsum(const float* __restrict__ colsum, float* __restrict__ affsum) {
  int t = blockIdx.x * blockDim.x + threadIdx.x;
  if (t >= BB * NS) return;
  int b = t >> 8, ij = t & 255, i = ij >> 4, j = ij & 15;
  float s = 0.f;
#pragma unroll
  for (int k = 0; k < 9; ++k) {
    int ii = i + 1 - (k / 3), jj = j + 1 - (k % 3);
    if (ii >= 0 && ii < GH && jj >= 0 && jj < GW)
      s += colsum[((size_t)b * NS + ii * GW + jj) * 16 + k];
  }
  affsum[t] = s;
}

__global__ void k_fold_st(const float* __restrict__ stn, const float* __restrict__ affsum,
                          _Float16* __restrict__ st2_h) {
  int t = blockIdx.x * blockDim.x + threadIdx.x;
  if (t >= BB * CC * NS) return;
  int ij = t & 255, c = (t >> 8) & 255, b = t >> 16;
  int i = ij >> 4, j = ij & 15;
  float s = 0.f;
#pragma unroll
  for (int k = 0; k < 9; ++k) {
    int ii = i + 1 - (k / 3), jj = j + 1 - (k % 3);
    if (ii >= 0 && ii < GH && jj >= 0 && jj < GW)
      s += stn[(((size_t)b * NS + ii * GW + jj) * CC + c) * 16 + k];
  }
  float a = affsum[(b << 8) | ij];
  st2_h[t] = (_Float16)(s / (a + 1e-12f));
}

// ------------------------------------------------------------------
// weight conversion f32 -> f16
// ------------------------------------------------------------------
__global__ void k_cvt_weights(const float* __restrict__ qkv_w, const float* __restrict__ proj_w,
                              _Float16* __restrict__ qkvw_h, _Float16* __restrict__ projw_h) {
  int t = blockIdx.x * blockDim.x + threadIdx.x;
  if (t < 768 * 256) qkvw_h[t] = (_Float16)qkv_w[t];
  if (t < 256 * 256) projw_h[t] = (_Float16)proj_w[t];
}

// ------------------------------------------------------------------
// K5/K7: tiled WMMA GEMM  O[b] = A(MxK) @ X[b](KxN) (+bias), f16 out.
// 64x64 block tile, 256 threads (8 waves), 2 subtiles per wave.
// ------------------------------------------------------------------
__global__ void k_gemm_wx(const _Float16* __restrict__ A, const _Float16* __restrict__ Xb,
                          const float* __restrict__ bias, _Float16* __restrict__ Ob,
                          int M, int N, int K, long strideX, long strideO) {
  __shared__ _Float16 As[64][32];   // [m][k]
  __shared__ _Float16 Bs[64][32];   // [n][k]
  const int m0 = blockIdx.x * 64, n0 = blockIdx.y * 64;
  const _Float16* X = Xb + (size_t)blockIdx.z * strideX;
  _Float16* O = Ob + (size_t)blockIdx.z * strideO;
  const int t = threadIdx.x;
  const int w = t >> 5, lane = t & 31;
  const int msub = w & 3, nbase = (w >> 2) * 2;
  v8f acc0 = {}, acc1 = {};
  for (int k0 = 0; k0 < K; k0 += 32) {
    {
      int row = t >> 2, c8 = (t & 3) * 8;
      const _Float16* src = A + (size_t)(m0 + row) * K + k0 + c8;
#pragma unroll
      for (int e = 0; e < 8; ++e) As[row][c8 + e] = src[e];
    }
    {
      int kk = t >> 3, n8 = (t & 7) * 8;
      const _Float16* src = X + (size_t)(k0 + kk) * N + n0 + n8;
#pragma unroll
      for (int e = 0; e < 8; ++e) Bs[n8 + e][kk] = src[e];
    }
    __syncthreads();
    v16h a  = frag_A16(&As[msub * 16][0], 32);
    v16h b0 = frag_B16(&Bs[nbase * 16][0], 32);
    v16h b1 = frag_B16(&Bs[(nbase + 1) * 16][0], 32);
    acc0 = WMMA_F32_F16(a, b0, acc0);
    acc1 = WMMA_F32_F16(a, b1, acc1);
    __syncthreads();
  }
  const int hi = (lane >> 4) ? 8 : 0, lo = lane & 15;
#pragma unroll
  for (int r = 0; r < 8; ++r) {
    int m = m0 + msub * 16 + hi + r;
    float bv = bias ? bias[m] : 0.f;
    O[(size_t)m * N + n0 + nbase * 16 + lo]       = (_Float16)(acc0[r] + bv);
    O[(size_t)m * N + n0 + (nbase + 1) * 16 + lo] = (_Float16)(acc1[r] + bv);
  }
}

// ------------------------------------------------------------------
// K5b: split qkv into WMMA-friendly global layouts:
//   qB[bh][m][d], kA[bh][n][d], vA[bh][d][n]   (bh = b2*8+head)
// ------------------------------------------------------------------
__global__ void k_qkv_split(const _Float16* __restrict__ qkv_h, _Float16* __restrict__ qB,
                            _Float16* __restrict__ kA, _Float16* __restrict__ vA) {
  int t = blockIdx.x * blockDim.x + threadIdx.x;
  if (t >= 16 * NAtt * HD) return;
  int d = t & 31, n = (t >> 5) & 1023, bh = t >> 15;
  int b2 = bh >> 3, head = bh & 7;
  int b = b2 * 4 + (n >> 8), hw = n & 255;
  const _Float16* src = qkv_h + (size_t)(b * 768 + head * 96) * 256 + hw;
  qB[t] = src[(size_t)d * 256];
  kA[t] = src[(size_t)(32 + d) * 256];
  vA[((size_t)bh * HD + d) * NAtt + n] = src[(size_t)(64 + d) * 256];
}

// ------------------------------------------------------------------
// K6: fused attention per (bh, 16-query tile).
//   S(1024x16) = K^T Q * scale (WMMA) -> 64KB f32 LDS slab,
//   column softmax over n (in place), O(32x16) = V @ P (WMMA, K split
//   across waves, cross-wave reduce through re-used slab memory).
// ------------------------------------------------------------------
__global__ void k_attention(const _Float16* __restrict__ qB, const _Float16* __restrict__ kA,
                            const _Float16* __restrict__ vA, _Float16* __restrict__ obuf_h) {
  __shared__ float slab[NAtt * 16];      // [n][16] 64KB, later reused for partials
  const int mt = blockIdx.x;             // 0..63
  const int bh = blockIdx.y;             // 0..15
  const int t = threadIdx.x;             // 256 threads, 8 waves
  const int w = t >> 5, lane = t & 31;
  const int hi = (lane >> 4) ? 8 : 0, lo = lane & 15;

  // phase 1: logits
  v16h bq = frag_B16(qB + ((size_t)bh * NAtt + mt * 16) * HD, HD);
  for (int nt = w * 8; nt < w * 8 + 8; ++nt) {
    v16h a = frag_A16(kA + ((size_t)bh * NAtt + nt * 16) * HD, HD);
    v8f acc = {};
    acc = WMMA_F32_F16(a, bq, acc);
#pragma unroll
    for (int r = 0; r < 8; ++r)
      slab[(nt * 16 + hi + r) * 16 + lo] = acc[r] * 0.17677669529663687f;  // hd^-0.5
  }
  __syncthreads();

  // phase 2: softmax over n for each column (16 threads per column)
  {
    const int col = t >> 4, l16 = t & 15;
    float m = -1e30f;
    for (int n = l16; n < NAtt; n += 16) m = fmaxf(m, slab[n * 16 + col]);
#pragma unroll
    for (int off = 8; off >= 1; off >>= 1) m = fmaxf(m, __shfl_xor(m, off, 16));
    float s = 0.f;
    for (int n = l16; n < NAtt; n += 16) s += __expf(slab[n * 16 + col] - m);
#pragma unroll
    for (int off = 8; off >= 1; off >>= 1) s += __shfl_xor(s, off, 16);
    float inv = 1.f / s;
    for (int n = l16; n < NAtt; n += 16)
      slab[n * 16 + col] = __expf(slab[n * 16 + col] - m) * inv;
  }
  __syncthreads();

  // phase 3: O = V * P   wave -> (dv-tile = w&1, K-chunk = w>>1)
  const int tile = w & 1, kpart = w >> 1;
  v8f oacc = {};
  for (int s = 0; s < 8; ++s) {
    int nn0 = (kpart * 8 + s) * 32;
    v16h a = frag_A16(vA + ((size_t)bh * HD + tile * 16) * NAtt + nn0, NAtt);
    v16h bp;
    {
      int koff = (lane >> 4) ? 16 : 0;
#pragma unroll
      for (int e = 0; e < 16; ++e)
        bp[e] = (_Float16)slab[(nn0 + koff + e) * 16 + lo];
    }
    oacc = WMMA_F32_F16(a, bp, oacc);
  }
  __syncthreads();                       // all slab reads done
  float* part = slab;                    // reuse LDS for partial accumulators
#pragma unroll
  for (int r = 0; r < 8; ++r) part[(w * 32 + lane) * 8 + r] = oacc[r];
  __syncthreads();
  if (t < 64) {
    const int rtile = t >> 5, rlane = t & 31;
    const int rhi = (rlane >> 4) ? 8 : 0, rlo = rlane & 15;
    const int mg = mt * 16 + rlo;
    const int b = (bh >> 3) * 4 + (mg >> 8);
    const int hw = mg & 255;
#pragma unroll
    for (int r = 0; r < 8; ++r) {
      float v = 0.f;
#pragma unroll
      for (int kp = 0; kp < 4; ++kp)
        v += part[((kp * 2 + rtile) * 32 + rlane) * 8 + r];
      int c = (bh & 7) * 32 + rtile * 16 + rhi + r;
      obuf_h[(size_t)(b * CC + c) * NS + hw] = (_Float16)v;
    }
  }
}

// ------------------------------------------------------------------
// K8: pix(Cx64) = sf2(Cx9, K padded to 32) @ affinity^T(9x64) per (b,n),
//     scattered to the (B,C,128,128) output layout.
// ------------------------------------------------------------------
__global__ void k_pixout(const _Float16* __restrict__ statt_h, const _Float16* __restrict__ aff_h,
                         float* __restrict__ out) {
  __shared__ _Float16 sfA[CC][32];    // [c][k]
  __shared__ _Float16 affB[PP][32];   // [p][k] (B operand [N][K])
  const int n = blockIdx.x, b = blockIdx.y;
  const int i = n >> 4, j = n & 15;
  const int t = threadIdx.x;
  {
    const int c = t;
#pragma unroll
    for (int k = 0; k < 32; ++k) {
      _Float16 v = (_Float16)0.f;
      if (k < 9) {
        int ni = i + (k / 3) - 1, nj = j + (k % 3) - 1;
        if (ni >= 0 && ni < GH && nj >= 0 && nj < GW)
          v = statt_h[(size_t)(b * CC + c) * NS + ni * GW + nj];
      }
      sfA[c][k] = v;
    }
  }
  if (t < PP) {
    const _Float16* src = aff_h + ((size_t)(b * NS + n) * PP + t) * 16;
#pragma unroll
    for (int k = 0; k < 16; ++k) affB[t][k] = src[k];
#pragma unroll
    for (int k = 16; k < 32; ++k) affB[t][k] = (_Float16)0.f;
  }
  __syncthreads();
  const int w = t >> 5, lane = t & 31;
  const int hi = (lane >> 4) ? 8 : 0, lo = lane & 15;
  for (int mtb = 0; mtb < 2; ++mtb) {
    const int mtile = w * 2 + mtb;
    v16h a = frag_A16(&sfA[mtile * 16][0], 32);
    for (int nt = 0; nt < 4; ++nt) {
      v16h bf = frag_B16(&affB[nt * 16][0], 32);
      v8f acc = {};
      acc = WMMA_F32_F16(a, bf, acc);
#pragma unroll
      for (int r = 0; r < 8; ++r) {
        int c = mtile * 16 + hi + r;
        int p = nt * 16 + lo;
        out[((size_t)(b * CC + c) * HH_ + i * 8 + (p >> 3)) * WW_ + j * 8 + (p & 7)] = acc[r];
      }
    }
  }
}

// ------------------------------------------------------------------
extern "C" void kernel_launch(void* const* d_in, const int* in_sizes, int n_in,
                              void* d_out, int out_size, void* d_ws, size_t ws_size,
                              hipStream_t stream) {
  const float* x      = (const float*)d_in[0];
  const float* qkv_w  = (const float*)d_in[1];
  const float* proj_w = (const float*)d_in[2];
  const float* proj_b = (const float*)d_in[3];
  float* out = (float*)d_out;

  char* ws = (char*)d_ws;
  size_t off = 0;
  auto alloc = [&](size_t bytes) -> char* {
    char* p = ws + off; off += (bytes + 255) & ~(size_t)255; return p;
  };
  _Float16* stoken_h = (_Float16*)alloc((size_t)BB * CC * NS * 2);
  _Float16* aff_h    = (_Float16*)alloc((size_t)BB * NS * PP * 16 * 2);
  float*    colsum   = (float*)   alloc((size_t)BB * NS * 16 * 4);
  float*    stn      = (float*)   alloc((size_t)BB * NS * CC * 16 * 4);
  float*    affsum   = (float*)   alloc((size_t)BB * NS * 4);
  _Float16* st2_h    = (_Float16*)alloc((size_t)BB * CC * NS * 2);
  _Float16* qkvw_h   = (_Float16*)alloc((size_t)768 * 256 * 2);
  _Float16* projw_h  = (_Float16*)alloc((size_t)256 * 256 * 2);
  _Float16* qkv_h    = (_Float16*)alloc((size_t)BB * 768 * 256 * 2);
  _Float16* qB       = (_Float16*)alloc((size_t)16 * NAtt * HD * 2);
  _Float16* kA       = (_Float16*)alloc((size_t)16 * NAtt * HD * 2);
  _Float16* vA       = (_Float16*)alloc((size_t)16 * NAtt * HD * 2);
  _Float16* obuf_h   = (_Float16*)alloc((size_t)BB * CC * NS * 2);
  _Float16* statt_h  = (_Float16*)alloc((size_t)BB * CC * NS * 2);

  k_stoken_mean<<<(BB * CC * NS) / 256, 256, 0, stream>>>(x, stoken_h);
  k_affinity<<<dim3(NS, BB), 128, 0, stream>>>(x, stoken_h, aff_h, colsum);
  k_stn<<<dim3(NS, BB), 256, 0, stream>>>(x, aff_h, stn);
  k_affsum<<<(BB * NS) / 256, 256, 0, stream>>>(colsum, affsum);
  k_fold_st<<<(BB * CC * NS) / 256, 256, 0, stream>>>(stn, affsum, st2_h);
  k_cvt_weights<<<(768 * 256) / 256, 256, 0, stream>>>(qkv_w, proj_w, qkvw_h, projw_h);
  k_gemm_wx<<<dim3(768 / 64, 256 / 64, BB), 256, 0, stream>>>(
      qkvw_h, st2_h, nullptr, qkv_h, 768, 256, 256, 256L * 256, 768L * 256);
  k_qkv_split<<<(16 * NAtt * HD) / 256, 256, 0, stream>>>(qkv_h, qB, kA, vA);
  k_attention<<<dim3(64, 16), 256, 0, stream>>>(qB, kA, vA, obuf_h);
  k_gemm_wx<<<dim3(256 / 64, 256 / 64, BB), 256, 0, stream>>>(
      projw_h, obuf_h, proj_b, statt_h, 256, 256, 256, 256L * 256, 256L * 256);
  k_pixout<<<dim3(NS, BB), 256, 0, stream>>>(statt_h, aff_h, out);

  (void)in_sizes; (void)n_in; (void)out_size; (void)ws_size;
}
